// SubGraph_46600395162188
// MI455X (gfx1250) — matrix operand order repaired and checked by reading, not compiled, not run on hardware
//
#include <hip/hip_runtime.h>
#include <hip/hip_bf16.h>

// ---------------------------------------------------------------------------
// Polyline encoder for MI455X (gfx1250, wave32, WMMA).
// One wave = one polyline (32 padded rows x 64 cols). All linears run through
// v_wmma_f32_16x16x32_f16 (2 M-tiles x 4 N-tiles, K unrolled). LayerNorm and
// max-pool over V are done in-register via __shfl_xor butterflies on the WMMA
// C-layout (VGPR r = row, lane = column, half = row-group).
// Weights are converted fp32 -> fp16 and pre-swizzled into B-fragment order
// (one contiguous 32B load per lane per fragment) by a pack prepass into d_ws.
// ---------------------------------------------------------------------------

typedef __attribute__((ext_vector_type(16))) _Float16 v16h;
typedef __attribute__((ext_vector_type(8)))  float    v8f;
static_assert(sizeof(v16h) == 32, "v16h must be 8 VGPRs");

#define NB    32
#define NP    256
#define NV    19
#define NCIN  19
#define NHID  64
#define NPOLY (NB * NP)

// f16-element offsets inside d_ws for the pre-swizzled B (= W^T) fragments.
// A weight with KT K-tiles occupies KT*4 tiles * 32 lanes * 16 halfs = KT*2048.
#define OFF_W1_0   0        // block0 w1  (K=19 -> pad 32, KT=1)
#define OFF_WSC_0  2048     // block0 wsc (KT=1)
#define OFF_W2_0   4096     // block0 w2  (K=64, KT=2)
#define OFF_W1_1   8192     // block1 w1  (K=128, KT=4)
#define OFF_WSC_1  16384
#define OFF_W2_1   24576
#define OFF_W1_2   28672
#define OFF_WSC_2  36864
#define OFF_W2_2   45056
#define OFF_WLIN   49152    // w_lin (K=128, KT=4)
#define WS_F16_TOTAL 57344  // 114,688 bytes

// ---------------------------------------------------------------------------
// Prepass: convert W[N=64][K] fp32 -> fragment-packed W^T fp16.
// B-fragment layout for v_wmma_*_16x16x32 (32x16 B, wave32):
//   lane: n = nt*16 + (lane&15); k-base = (lane<16 ? 0 : 8)
//   element e (0..15): k = ktile*32 + kbase + (e&7) + (e>=8 ? 16 : 0)
// Packed so that each lane reads 16 contiguous f16 (32B) per fragment.
// ---------------------------------------------------------------------------
__global__ void pack_b_frag(const float* __restrict__ src, int K,
                            _Float16* __restrict__ dst, int nelem) {
  int idx = blockIdx.x * blockDim.x + threadIdx.x;
  if (idx >= nelem) return;
  int tile   = idx >> 9;          // (kt*4 + nt)
  int within = idx & 511;
  int lane   = within >> 4;
  int e      = within & 15;
  int kt = tile >> 2, nt = tile & 3;
  int n = nt * 16 + (lane & 15);
  int k = kt * 32 + ((lane >> 4) << 3) + (e & 7) + ((e >> 3) << 4);
  dst[idx] = (k < K) ? (_Float16)src[n * K + k] : (_Float16)0.0f;
}

// ---------------------------------------------------------------------------
// Device helpers
// ---------------------------------------------------------------------------
struct BlkP  { const float *b1,*g1,*be1,*b2,*g2,*be2,*bsc,*gsc,*besc; };
struct TailP { const float *b_lin,*g_out,*be_out,*w_out,*b_out; };

__device__ __forceinline__ v8f wmma_f16(v16h a, v16h b, v8f c) {
  return __builtin_amdgcn_wmma_f32_16x16x32_f16(
      /*neg_a=*/false, a, /*neg_b=*/false, b,
      /*c_mod=*/(short)0, c, /*reuse_a=*/false, /*reuse_b=*/false);
}

__device__ __forceinline__ void zero_acc(v8f a[2][4]) {
#pragma unroll
  for (int i = 0; i < 2; ++i)
#pragma unroll
    for (int j = 0; j < 4; ++j)
#pragma unroll
      for (int k = 0; k < 8; ++k) a[i][j][k] = 0.0f;
}

// A-fragment (16x32 f16) from LDS, row-major with stride lda.
// ISA layout: lanes 0-15: row=lane, VGPR0-3: K 0..7, VGPR4-7: K 16..23.
//             lanes 16-31: row=lane-16, K 8..15 and 24..31.
__device__ __forceinline__ v16h load_a_frag(const _Float16* s, int lda, int lane) {
  union { unsigned u[8]; v16h h; } r;
  int m  = lane & 15;
  int kb = (lane >> 4) << 3;            // 0 or 8
  const _Float16* row = s + m * lda + kb;
#pragma unroll
  for (int j = 0; j < 4; ++j) {
    r.u[j]     = *(const unsigned*)(row + 2 * j);        // K = kb + 2j, 2j+1
    r.u[4 + j] = *(const unsigned*)(row + 16 + 2 * j);   // K = kb + 16 + 2j, +1
  }
  return r.h;
}

// 32x64 GEMM: acc += A(32xK, LDS f16) * B(Kx64, pre-swizzled ws f16). K = KT*32.
__device__ __forceinline__ void gemm32x64(const _Float16* As, int lda,
                                          const _Float16* Bw, int KT, int lane,
                                          v8f acc[2][4]) {
#pragma unroll
  for (int kt = 0; kt < KT; ++kt) {
    v16h a0 = load_a_frag(As + kt * 32, lda, lane);
    v16h a1 = load_a_frag(As + 16 * lda + kt * 32, lda, lane);
#pragma unroll
    for (int nt = 0; nt < 4; ++nt) {
      v16h b = *(const v16h*)(Bw + (size_t)(kt * 4 + nt) * 512 + lane * 16);
      acc[0][nt] = wmma_f16(a0, b, acc[0][nt]);
      acc[1][nt] = wmma_f16(a1, b, acc[1][nt]);
    }
  }
}

// Row statistics over 64 channels held by the 16 lanes of this half-wave.
__device__ __forceinline__ void row_stats(float s, float q, float& mean, float& rstd) {
#pragma unroll
  for (int off = 1; off < 16; off <<= 1) {
    s += __shfl_xor(s, off, 32);
    q += __shfl_xor(q, off, 32);
  }
  mean = s * (1.0f / 64.0f);
  float var = q * (1.0f / 64.0f) - mean * mean;
  rstd = rsqrtf(var + 1e-5f);
}

// LN(acc + bias)*g + be (optional ReLU) -> f16 LDS [32 x ldd].
__device__ __forceinline__ void ln_store(const v8f acc[2][4],
                                         const float* bias, const float* g, const float* be,
                                         _Float16* dst, int ldd, int lane, bool relu) {
  int col = lane & 15, half = lane >> 4;
  float bv[4], gv[4], bev[4];
#pragma unroll
  for (int nt = 0; nt < 4; ++nt) {
    int ch = nt * 16 + col;
    bv[nt] = bias[ch]; gv[nt] = g[ch]; bev[nt] = be[ch];
  }
#pragma unroll
  for (int mt = 0; mt < 2; ++mt)
#pragma unroll
    for (int r = 0; r < 8; ++r) {
      float v[4], s = 0.f, q = 0.f;
#pragma unroll
      for (int nt = 0; nt < 4; ++nt) {
        v[nt] = acc[mt][nt][r] + bv[nt];
        s += v[nt]; q += v[nt] * v[nt];
      }
      float mean, rstd; row_stats(s, q, mean, rstd);
      int row = mt * 16 + half * 8 + r;
#pragma unroll
      for (int nt = 0; nt < 4; ++nt) {
        float o = (v[nt] - mean) * rstd * gv[nt] + bev[nt];
        if (relu) o = fmaxf(o, 0.0f);
        dst[row * ldd + nt * 16 + col] = (_Float16)o;
      }
    }
}

// y = relu(LN(h2)+LN(sc)); write y -> sX[:, 0:64]; agg=max over valid rows -> sX[:, 64:128].
__device__ __forceinline__ void block_tail(const v8f ah[2][4], const v8f as[2][4],
                                           const BlkP& P, _Float16* sXw, int lane) {
  int col = lane & 15, half = lane >> 4;
  float b2v[4], g2v[4], be2v[4], bsv[4], gsv[4], besv[4];
#pragma unroll
  for (int nt = 0; nt < 4; ++nt) {
    int ch = nt * 16 + col;
    b2v[nt] = P.b2[ch];  g2v[nt] = P.g2[ch];  be2v[nt] = P.be2[ch];
    bsv[nt] = P.bsc[ch]; gsv[nt] = P.gsc[ch]; besv[nt] = P.besc[ch];
  }
  float mx[4] = {-1e30f, -1e30f, -1e30f, -1e30f};
#pragma unroll
  for (int mt = 0; mt < 2; ++mt)
#pragma unroll
    for (int r = 0; r < 8; ++r) {
      float vh[4], vs[4], sh = 0.f, qh = 0.f, ss = 0.f, qs = 0.f;
#pragma unroll
      for (int nt = 0; nt < 4; ++nt) {
        vh[nt] = ah[mt][nt][r] + b2v[nt]; sh += vh[nt]; qh += vh[nt] * vh[nt];
        vs[nt] = as[mt][nt][r] + bsv[nt]; ss += vs[nt]; qs += vs[nt] * vs[nt];
      }
      float mh, rh, ms, rs;
      row_stats(sh, qh, mh, rh);
      row_stats(ss, qs, ms, rs);
      int row = mt * 16 + half * 8 + r;
#pragma unroll
      for (int nt = 0; nt < 4; ++nt) {
        float y = fmaxf((vh[nt] - mh) * rh * g2v[nt] + be2v[nt] +
                        (vs[nt] - ms) * rs * gsv[nt] + besv[nt], 0.0f);
        sXw[row * 128 + nt * 16 + col] = (_Float16)y;
        if (row < NV) mx[nt] = fmaxf(mx[nt], y);
      }
    }
#pragma unroll
  for (int nt = 0; nt < 4; ++nt)
    mx[nt] = fmaxf(mx[nt], __shfl_xor(mx[nt], 16, 32));
  // broadcast agg into columns 64..127 for all 32 rows (each half writes 16 rows)
#pragma unroll
  for (int nt = 0; nt < 4; ++nt) {
    _Float16 hv = (_Float16)mx[nt];
#pragma unroll
    for (int rr = 0; rr < 16; ++rr)
      sXw[(half * 16 + rr) * 128 + 64 + nt * 16 + col] = hv;
  }
}

__device__ __forceinline__ void run_block(const _Float16* w1p, const _Float16* w2p,
                                          const _Float16* wscp, int KT, const BlkP& P,
                                          _Float16* sXw, _Float16* sHw, int lane) {
  v8f acc[2][4]; zero_acc(acc);
  gemm32x64(sXw, 128, w1p, KT, lane, acc);
  ln_store(acc, P.b1, P.g1, P.be1, sHw, 64, lane, /*relu=*/true);
  __syncthreads();
  v8f ah[2][4]; zero_acc(ah);
  gemm32x64(sHw, 64, w2p, 2, lane, ah);
  v8f as[2][4]; zero_acc(as);
  gemm32x64(sXw, 128, wscp, KT, lane, as);
  __syncthreads();
  block_tail(ah, as, P, sXw, lane);
  __syncthreads();
}

// ---------------------------------------------------------------------------
// Main fused kernel: 128 threads = 4 waves = 4 polylines per workgroup.
// ---------------------------------------------------------------------------
__global__ void __launch_bounds__(128)
polyline_encoder(const float* __restrict__ feats,
                 const unsigned char* __restrict__ mask,
                 const float* __restrict__ attr,
                 const _Float16* __restrict__ ws,
                 BlkP P0, BlkP P1, BlkP P2, TailP T,
                 float* __restrict__ out) {
  __shared__ _Float16 sX[4][32 * 128];  // activation (y | agg), fp16
  __shared__ _Float16 sH[4][32 * 64];   // hidden after LN1+ReLU, fp16
  __shared__ float    sV[4][64];        // final per-polyline vector

  int w = threadIdx.x >> 5, lane = threadIdx.x & 31;
  int poly = blockIdx.x * 4 + w;
  _Float16* sXw = sX[w];
  _Float16* sHw = sH[w];
  float*    sVw = sV[w];

  // Load feats[poly] (19x19) into sX cols 0..31, zero-padded to 32x32.
  const float* f = feats + (size_t)poly * NV * NCIN;
  for (int t = lane; t < 32 * 32; t += 32) {
    int r = t >> 5, c = t & 31;
    float v = (r < NV && c < NCIN) ? f[r * NCIN + c] : 0.0f;
    sXw[r * 128 + c] = (_Float16)v;
  }
  __syncthreads();

  run_block(ws + OFF_W1_0, ws + OFF_W2_0, ws + OFF_WSC_0, 1, P0, sXw, sHw, lane);
  run_block(ws + OFF_W1_1, ws + OFF_W2_1, ws + OFF_WSC_1, 4, P1, sXw, sHw, lane);
  run_block(ws + OFF_W1_2, ws + OFF_W2_2, ws + OFF_WSC_2, 4, P2, sXw, sHw, lane);

  // Final linear (K=128) + max over V + attr-sum + LN + ReLU.
  v8f acc[2][4]; zero_acc(acc);
  gemm32x64(sXw, 128, ws + OFF_WLIN, 4, lane, acc);

  int col = lane & 15, half = lane >> 4;
  float blv[4], mx[4] = {-1e30f, -1e30f, -1e30f, -1e30f};
#pragma unroll
  for (int nt = 0; nt < 4; ++nt) blv[nt] = T.b_lin[nt * 16 + col];
#pragma unroll
  for (int mt = 0; mt < 2; ++mt)
#pragma unroll
    for (int r = 0; r < 8; ++r) {
      int row = mt * 16 + half * 8 + r;
      if (row < NV)
#pragma unroll
        for (int nt = 0; nt < 4; ++nt)
          mx[nt] = fmaxf(mx[nt], acc[mt][nt][r] + blv[nt]);
    }
#pragma unroll
  for (int nt = 0; nt < 4; ++nt)
    mx[nt] = fmaxf(mx[nt], __shfl_xor(mx[nt], 16, 32));

  const float* a0 = attr + (size_t)poly * NHID;
  const float* a1 = attr + (size_t)(NPOLY + poly) * NHID;
  float s = 0.f, q = 0.f;
#pragma unroll
  for (int nt = 0; nt < 4; ++nt) {
    int ch = nt * 16 + col;
    mx[nt] += a0[ch] + a1[ch];
    s += mx[nt]; q += mx[nt] * mx[nt];
  }
  float mean, rstd; row_stats(s, q, mean, rstd);
#pragma unroll
  for (int nt = 0; nt < 4; ++nt) {
    int ch = nt * 16 + col;
    float o = fmaxf((mx[nt] - mean) * rstd * T.g_out[ch] + T.be_out[ch], 0.0f);
    if (half == 0) sVw[ch] = o;  // halves hold identical values
  }
  __syncthreads();

  // out = vec @ w_out.T + b_out ; each lane owns channels (lane, lane+32)
  float o0 = T.b_out[lane], o1 = T.b_out[lane + 32];
  for (int k = 0; k < 64; ++k) {
    float vk = sVw[k];
    o0 = fmaf(vk, T.w_out[lane * 64 + k], o0);
    o1 = fmaf(vk, T.w_out[(lane + 32) * 64 + k], o1);
  }
  float mk = mask[poly] ? 1.0f : 0.0f;
  out[(size_t)poly * NHID + lane]      = o0 * mk;
  out[(size_t)poly * NHID + lane + 32] = o1 * mk;
}

// ---------------------------------------------------------------------------
// Host launch. Input order = jax tree-flatten (dict keys sorted):
//  0 feats, 1 mask(bool), 2 attr, 3 b_lin, 4 b_out, 5 be_out,
//  6..41 blocks[0..2] each {b1,b2,be1,be2,besc,bsc,g1,g2,gsc,w1,w2,wsc},
//  42 g_out, 43 w_lin, 44 w_out.
// ---------------------------------------------------------------------------
extern "C" void kernel_launch(void* const* d_in, const int* in_sizes, int n_in,
                              void* d_out, int out_size, void* d_ws, size_t ws_size,
                              hipStream_t stream) {
  const float*         feats  = (const float*)d_in[0];
  const unsigned char* mask   = (const unsigned char*)d_in[1];
  const float*         attr   = (const float*)d_in[2];
  const float*         b_lin  = (const float*)d_in[3];
  const float*         b_out  = (const float*)d_in[4];
  const float*         be_out = (const float*)d_in[5];
  const float* blk[3][12];
  for (int i = 0; i < 3; ++i)
    for (int j = 0; j < 12; ++j)
      blk[i][j] = (const float*)d_in[6 + i * 12 + j];
  const float* g_out = (const float*)d_in[42];
  const float* w_lin = (const float*)d_in[43];
  const float* w_out = (const float*)d_in[44];

  _Float16* ws = (_Float16*)d_ws;

  auto pk = [&](const float* W, int K, int off, int KT) {
    int n = KT * 2048;
    pack_b_frag<<<(n + 255) / 256, 256, 0, stream>>>(W, K, ws + off, n);
  };
  // blk idx: 0 b1,1 b2,2 be1,3 be2,4 besc,5 bsc,6 g1,7 g2,8 gsc,9 w1,10 w2,11 wsc
  pk(blk[0][9],  NCIN, OFF_W1_0,  1);
  pk(blk[0][11], NCIN, OFF_WSC_0, 1);
  pk(blk[0][10], NHID, OFF_W2_0,  2);
  pk(blk[1][9],  128,  OFF_W1_1,  4);
  pk(blk[1][11], 128,  OFF_WSC_1, 4);
  pk(blk[1][10], NHID, OFF_W2_1,  2);
  pk(blk[2][9],  128,  OFF_W1_2,  4);
  pk(blk[2][11], 128,  OFF_WSC_2, 4);
  pk(blk[2][10], NHID, OFF_W2_2,  2);
  pk(w_lin,      128,  OFF_WLIN,  4);

  BlkP P[3];
  for (int i = 0; i < 3; ++i)
    P[i] = BlkP{blk[i][0], blk[i][6], blk[i][2],   // b1, g1, be1
                blk[i][1], blk[i][7], blk[i][3],   // b2, g2, be2
                blk[i][5], blk[i][8], blk[i][4]};  // bsc, gsc, besc
  TailP T{b_lin, g_out, be_out, w_out, b_out};

  polyline_encoder<<<NPOLY / 4, 128, 0, stream>>>(
      feats, mask, attr, ws, P[0], P[1], P[2], T, (float*)d_out);
}
